// Aggregator_39281770889636
// MI455X (gfx1250) — compile-verified, matrix operand order checked by reading, not run on hardware
//
#include <hip/hip_runtime.h>
#include <hip/hip_bf16.h>
#include <math.h>

typedef __attribute__((ext_vector_type(16))) _Float16 v16h;
typedef __attribute__((ext_vector_type(8)))  float    v8f;

union FragAB { v16h v; uint4 q[2]; };

__device__ __forceinline__ float lrelu(float x) { return x > 0.f ? x : 0.01f * x; }

// ---------------------------------------------------------------------------
// Single row-tile GEMM: C(16x16) += A(16xK) * B(KxN tile) over kChunks of K=32.
// A row-major (lda halves), B is W^T row-major (row n, col k; ldb halves), so
// both use the ISA 16-bit A fragment pattern (lane = 16h + m; halves 0-7 =
// K[k0+8h..+7], halves 8-15 = K[k0+16+8h..+7]).
// ---------------------------------------------------------------------------
__device__ __forceinline__ v8f wmma_k(const _Float16* __restrict__ A, int lda,
                                      const _Float16* __restrict__ B, int ldb,
                                      int kChunks, v8f acc, int lane)
{
  const int m = lane & 15;
  const int h = lane >> 4;
  for (int kc = 0; kc < kChunks; ++kc) {
    const int k0 = kc * 32 + h * 8;
    FragAB a, b;
    a.q[0] = *(const uint4*)(A + (size_t)m * lda + k0);
    a.q[1] = *(const uint4*)(A + (size_t)m * lda + k0 + 16);
    b.q[0] = *(const uint4*)(B + (size_t)m * ldb + k0);
    b.q[1] = *(const uint4*)(B + (size_t)m * ldb + k0 + 16);
    acc = __builtin_amdgcn_wmma_f32_16x16x32_f16(false, a.v, false, b.v,
                                                 (short)0, acc, false, false);
  }
  return acc;
}

// ---------------------------------------------------------------------------
// Multi-row-tile GEMM: 8 row tiles (128 rows) share each B fragment, so every
// weight fetch feeds 8 WMMAs -> 8x less L2 weight traffic than 16-row blocks.
// ---------------------------------------------------------------------------
__device__ __forceinline__ void gemm_mt8(const _Float16* __restrict__ A, int lda,
                                         const _Float16* __restrict__ Bt, int ldb,
                                         int kChunks, v8f* acc, int lane)
{
  const int m = lane & 15;
  const int h = lane >> 4;
  for (int kc = 0; kc < kChunks; ++kc) {
    const int k0 = kc * 32 + h * 8;
    FragAB b;
    const _Float16* Bp = Bt + (size_t)m * ldb + k0;
    b.q[0] = *(const uint4*)(Bp);
    b.q[1] = *(const uint4*)(Bp + 16);
    if (kc + 1 < kChunks) __builtin_prefetch(Bp + 32, 0, 1); // global_prefetch_b8
    #pragma unroll
    for (int rt = 0; rt < 8; ++rt) {
      FragAB a;
      const _Float16* Ar = A + (size_t)(rt * 16 + m) * lda + k0;
      a.q[0] = *(const uint4*)(Ar);
      a.q[1] = *(const uint4*)(Ar + 16);
      acc[rt] = __builtin_amdgcn_wmma_f32_16x16x32_f16(false, a.v, false, b.v,
                                                       (short)0, acc[rt], false, false);
    }
  }
}

// Epilogue: bias + leaky-relu, f16 store. Acc layout: VGPR v holds row
// (v + 8*(lane>=16)), column (lane&15) within a 16x16 tile.
__device__ __forceinline__ void store_tile_f16(v8f acc, const float* __restrict__ bias,
                                               int col0, _Float16* dst, int ldd, int lane)
{
  const int col = col0 + (lane & 15);
  const float bf = bias[col];
  const int rbase = (lane >> 4) * 8;
  #pragma unroll
  for (int v = 0; v < 8; ++v) {
    float x = lrelu(acc[v] + bf);
    dst[(size_t)(rbase + v) * ldd + col] = (_Float16)x;
  }
}

__device__ __forceinline__ void store_mt8_f16(const v8f* acc, const float* __restrict__ bias,
                                              int col0, _Float16* dst, int ldd, int lane)
{
  const int col = col0 + (lane & 15);
  const float bf = bias[col];
  const int rb = (lane >> 4) * 8;
  #pragma unroll
  for (int rt = 0; rt < 8; ++rt) {
    #pragma unroll
    for (int v = 0; v < 8; ++v) {
      float x = lrelu(acc[rt][v] + bf);
      dst[(size_t)(rt * 16 + rb + v) * ldd + col] = (_Float16)x;
    }
  }
}

// ---------------------------------------------------------------------------
// Weight prep: f32 (K x N) -> f16 W^T (N x Kpad), zero padded K -> Kpad.
// ---------------------------------------------------------------------------
__global__ void wprep_kernel(const float* __restrict__ src, _Float16* __restrict__ dst,
                             int K, int N, int Kpad)
{
  int idx = blockIdx.x * blockDim.x + threadIdx.x;
  if (idx >= N * Kpad) return;
  int n  = idx / Kpad;
  int kk = idx - n * Kpad;
  float v = (kk < K) ? src[(size_t)kk * N + n] : 0.0f;
  dst[idx] = (_Float16)v;
}

// ---------------------------------------------------------------------------
// Kernel 1: per-point MLP + aggregation.
// Block = 256 thr (8 waves), 128 rows = 8 samples. 4096 blocks.
// Each wave: 2 column tiles x 8 row-tile accumulators.
// ---------------------------------------------------------------------------
#define ROWS 128
#define NPB  8

__global__ void __launch_bounds__(256)
point_mlp_kernel(const float* __restrict__ pc_xyz, const float* __restrict__ pc_rgb,
                 const float* __restrict__ pc_dir, const float* __restrict__ pc_feat,
                 const float* __restrict__ sp_xyz, const unsigned char* __restrict__ vmask,
                 const float* __restrict__ rays_d,
                 const _Float16* __restrict__ Wt1, const float* __restrict__ b1,
                 const _Float16* __restrict__ Wt2, const float* __restrict__ b2,
                 const _Float16* __restrict__ Wt3, const float* __restrict__ b3,
                 const _Float16* __restrict__ Wt4, const float* __restrict__ b4,
                 const float* __restrict__ wa, const float* __restrict__ ba,
                 _Float16* __restrict__ feat_out /* [N][288] f16 */,
                 float* __restrict__ alpha_out /* [N] */)
{
  const int n0   = blockIdx.x * NPB;  // first sample of this block
  const int tid  = threadIdx.x;       // 0..255
  const int wave = tid >> 5;          // 0..7
  const int lane = tid & 31;

  extern __shared__ char smem[];
  _Float16* xin  = (_Float16*)smem;                       // ROWS x 288
  _Float16* hbuf = (_Float16*)(smem + ROWS * 288 * 2);    // ROWS x 256

  __shared__ float sh_df[ROWS][6];   // dist(3) + distpow(3)
  __shared__ float sh_inv[ROWS];
  __shared__ float wrow[ROWS];       // normalized interp weight per row
  __shared__ float apart[ROWS * 2];  // alpha partial dots
  __shared__ float salpha[ROWS];

  // ---- per-row geometry + inverse-distance weights --------------------------
  if (tid < ROWS) {
    int r = tid;
    int ns = n0 + (r >> 4);
    size_t pb = ((size_t)n0 * 16 + r) * 3;   // rows are contiguous across samples
    float x0 = pc_xyz[pb], x1 = pc_xyz[pb + 1], x2 = pc_xyz[pb + 2];
    float s0 = sp_xyz[(size_t)ns * 3], s1 = sp_xyz[(size_t)ns * 3 + 1], s2 = sp_xyz[(size_t)ns * 3 + 2];
    float d0 = x0 - s0, d1 = x1 - s1, d2 = x2 - s2;
    sh_df[r][0] = d0; sh_df[r][1] = d1; sh_df[r][2] = d2;
    sh_df[r][3] = x0 * x0 - s0 * s0;
    sh_df[r][4] = x1 * x1 - s1 * s1;
    sh_df[r][5] = x2 * x2 - s2 * s2;
    float dn = sqrtf(d0 * d0 + d1 * d1 + d2 * d2);
    float mk = vmask[(size_t)n0 * 16 + r] ? 1.0f : 0.0f;
    sh_inv[r] = (1.0f / (dn + 1e-6f)) * mk;
  }
  __syncthreads();
  if (tid < NPB) {
    int g = tid;
    float s = 0.f;
    #pragma unroll
    for (int r = 0; r < 16; ++r) s += sh_inv[g * 16 + r];
    float is = 1.0f / (s + 1e-6f);
    #pragma unroll
    for (int r = 0; r < 16; ++r) wrow[g * 16 + r] = sh_inv[g * 16 + r] * is;
  }

  // ---- build 284-wide encoded input (padded to 288) -------------------------
  // [pc_feat 32 | emb_feat 192 (j*3+f, sin/cos) | emb_dist 60 (j*5+f) | 0 pad]
  for (int idx = tid; idx < ROWS * 288; idx += 256) {
    int r = idx / 288, c = idx - r * 288;
    float v;
    if (c < 32) {
      v = pc_feat[((size_t)n0 * 16 + r) * 32 + c];
    } else if (c < 224) {
      int e = c - 32, p = e >> 1, j = p / 3, f = p - 3 * j;
      float base = pc_feat[((size_t)n0 * 16 + r) * 32 + j] * (float)(1 << f);
      v = (e & 1) ? __cosf(base) : __sinf(base);
    } else if (c < 284) {
      int e = c - 224, p = e >> 1, j = p / 5, f = p - 5 * j;
      float base = sh_df[r][j] * (float)(1 << f);
      v = (e & 1) ? __cosf(base) : __sinf(base);
    } else {
      v = 0.0f;
    }
    xin[idx] = (_Float16)v;
  }
  __syncthreads();

  // ---- L1: 288 -> 256 -------------------------------------------------------
  for (int t = wave; t < 16; t += 8) {
    v8f acc[8];
    #pragma unroll
    for (int i = 0; i < 8; ++i) acc[i] = v8f{0.f, 0.f, 0.f, 0.f, 0.f, 0.f, 0.f, 0.f};
    gemm_mt8(xin, 288, Wt1 + (size_t)t * 16 * 288, 288, 9, acc, lane);
    store_mt8_f16(acc, b1, t * 16, hbuf, 256, lane);
  }
  __syncthreads();

  // ---- L2: 256 -> 256 (into xin cols 0..255) --------------------------------
  for (int t = wave; t < 16; t += 8) {
    v8f acc[8];
    #pragma unroll
    for (int i = 0; i < 8; ++i) acc[i] = v8f{0.f, 0.f, 0.f, 0.f, 0.f, 0.f, 0.f, 0.f};
    gemm_mt8(hbuf, 256, Wt2 + (size_t)t * 16 * 256, 256, 8, acc, lane);
    store_mt8_f16(acc, b2, t * 16, xin, 288, lane);
  }
  // extras: cols 256..258 rgb, 259..261 dir-ray, 262 dot, 263..287 zero
  for (int idx = tid; idx < ROWS * 32; idx += 256) {
    int r = idx >> 5, c = idx & 31;
    int ns = n0 + (r >> 4);
    float v = 0.0f;
    size_t pb = ((size_t)n0 * 16 + r) * 3;
    if (c < 3) {
      v = pc_rgb[pb + c];
    } else if (c < 6) {
      v = pc_dir[pb + (c - 3)] - rays_d[(size_t)ns * 3 + (c - 3)];
    } else if (c == 6) {
      v = pc_dir[pb] * rays_d[(size_t)ns * 3] +
          pc_dir[pb + 1] * rays_d[(size_t)ns * 3 + 1] +
          pc_dir[pb + 2] * rays_d[(size_t)ns * 3 + 2];
    }
    xin[(size_t)r * 288 + 256 + c] = (_Float16)v;
  }
  __syncthreads();

  // ---- L3: 288(263) -> 256 --------------------------------------------------
  for (int t = wave; t < 16; t += 8) {
    v8f acc[8];
    #pragma unroll
    for (int i = 0; i < 8; ++i) acc[i] = v8f{0.f, 0.f, 0.f, 0.f, 0.f, 0.f, 0.f, 0.f};
    gemm_mt8(xin, 288, Wt3 + (size_t)t * 16 * 288, 288, 9, acc, lane);
    store_mt8_f16(acc, b3, t * 16, hbuf, 256, lane);
  }
  __syncthreads();

  // ---- L4: 256 -> 256 (f16 out into xin cols 0..255) ------------------------
  for (int t = wave; t < 16; t += 8) {
    v8f acc[8];
    #pragma unroll
    for (int i = 0; i < 8; ++i) acc[i] = v8f{0.f, 0.f, 0.f, 0.f, 0.f, 0.f, 0.f, 0.f};
    gemm_mt8(hbuf, 256, Wt4 + (size_t)t * 16 * 256, 256, 8, acc, lane);
    store_mt8_f16(acc, b4, t * 16, xin, 288, lane);
  }
  __syncthreads();

  // ---- alpha: softplus(feat . wa + ba - 1), weighted sum per sample ---------
  {
    // stage 1: 256 thr = 8 samples x (16 rows x 2 halves)
    int g = tid >> 5, s = tid & 31, r = s >> 1, half = s & 1;
    int row = g * 16 + r;
    float p = 0.f;
    for (int c = half * 128; c < half * 128 + 128; ++c)
      p += (float)xin[(size_t)row * 288 + c] * wa[c];
    apart[row * 2 + half] = p;
  }
  __syncthreads();
  if (tid < ROWS) {
    float d = apart[tid * 2] + apart[tid * 2 + 1] + ba[0] - 1.0f;
    float sp = (d > 20.f) ? d : log1pf(__expf(d));
    salpha[tid] = sp * wrow[tid];
  }
  __syncthreads();
  if (tid < NPB) {
    float a = 0.f;
    #pragma unroll
    for (int r = 0; r < 16; ++r) a += salpha[tid * 16 + r];
    alpha_out[n0 + tid] = a;
  }

  // ---- feat_agg (256) + emb_ray (24) + pad -> feat_out[ns][288] -------------
  {
    int c = tid;  // 0..255
    #pragma unroll
    for (int g = 0; g < NPB; ++g) {
      float ag = 0.f;
      #pragma unroll
      for (int r = 0; r < 16; ++r)
        ag += (float)xin[(size_t)(g * 16 + r) * 288 + c] * wrow[g * 16 + r];
      feat_out[(size_t)(n0 + g) * 288 + c] = (_Float16)ag;
    }
  }
  {
    int g = tid >> 5, e = tid & 31;  // 8 samples x 32 tail cols
    float v = 0.0f;
    if (e < 24) {
      int p = e >> 1, j = p >> 2, f = p & 3;
      float base = rays_d[(size_t)(n0 + g) * 3 + j] * (float)(1 << f);
      v = (e & 1) ? __cosf(base) : __sinf(base);
    }
    feat_out[(size_t)(n0 + g) * 288 + 256 + e] = (_Float16)v;
  }
}

// ---------------------------------------------------------------------------
// Kernel 2: ray color head. 16 rows / block (128 thr = 4 waves), 2048 blocks.
// ---------------------------------------------------------------------------
__global__ void __launch_bounds__(128)
ray_head_kernel(const _Float16* __restrict__ feat_in /* [N][288] */,
                const _Float16* __restrict__ Wc1, const float* __restrict__ bc1,
                const _Float16* __restrict__ Wc2, const float* __restrict__ bc2,
                const _Float16* __restrict__ Wc3, const float* __restrict__ bc3,
                const float* __restrict__ wc4, const float* __restrict__ bc4,
                float* __restrict__ color_out)
{
  const int n0   = blockIdx.x * 16;
  const int tid  = threadIdx.x;
  const int wave = tid >> 5;
  const int lane = tid & 31;

  __shared__ _Float16 hA[16 * 128];
  __shared__ _Float16 hB[16 * 128];

  const _Float16* A0 = feat_in + (size_t)n0 * 288;
  for (int t = wave; t < 8; t += 4) {
    v8f acc = {0.f, 0.f, 0.f, 0.f, 0.f, 0.f, 0.f, 0.f};
    acc = wmma_k(A0, 288, Wc1 + (size_t)t * 16 * 288, 288, 9, acc, lane);
    store_tile_f16(acc, bc1, t * 16, hA, 128, lane);
  }
  __syncthreads();

  for (int t = wave; t < 8; t += 4) {
    v8f acc = {0.f, 0.f, 0.f, 0.f, 0.f, 0.f, 0.f, 0.f};
    acc = wmma_k(hA, 128, Wc2 + (size_t)t * 16 * 128, 128, 4, acc, lane);
    store_tile_f16(acc, bc2, t * 16, hB, 128, lane);
  }
  __syncthreads();

  for (int t = wave; t < 8; t += 4) {
    v8f acc = {0.f, 0.f, 0.f, 0.f, 0.f, 0.f, 0.f, 0.f};
    acc = wmma_k(hB, 128, Wc3 + (size_t)t * 16 * 128, 128, 4, acc, lane);
    store_tile_f16(acc, bc3, t * 16, hA, 128, lane);
  }
  __syncthreads();

  if (tid < 48) {
    int r = tid / 3, cc = tid - 3 * r;
    float s = bc4[cc];
    for (int k = 0; k < 128; ++k) s += (float)hA[(size_t)r * 128 + k] * wc4[(size_t)k * 3 + cc];
    float sg = 1.0f / (1.0f + __expf(-s));
    color_out[(size_t)(n0 + r) * 3 + cc] = sg * 1.002f - 0.001f;
  }
}

// ---------------------------------------------------------------------------
// Workspace layout (bytes)
// ---------------------------------------------------------------------------
#define WS_WT1   0u          // 256 x 288 f16 = 147456
#define WS_WT2   147456u     // 256 x 256 f16 = 131072
#define WS_WT3   278528u     // 256 x 288 f16 = 147456
#define WS_WT4   425984u     // 256 x 256 f16 = 131072
#define WS_WTC1  557056u     // 128 x 288 f16 = 73728
#define WS_WTC2  630784u     // 128 x 128 f16 = 32768
#define WS_WTC3  663552u     // 128 x 128 f16 = 32768
#define WS_FEAT  696320u     // 32768 x 288 f16 = 18874368

extern "C" void kernel_launch(void* const* d_in, const int* in_sizes, int n_in,
                              void* d_out, int out_size, void* d_ws, size_t ws_size,
                              hipStream_t stream) {
  (void)in_sizes; (void)n_in; (void)out_size; (void)ws_size;

  const float* pc_xyz = (const float*)d_in[0];
  const float* pc_rgb = (const float*)d_in[1];
  const float* pc_dir = (const float*)d_in[2];
  const float* pc_feat = (const float*)d_in[3];
  const float* sp_xyz = (const float*)d_in[4];
  const unsigned char* vmask = (const unsigned char*)d_in[5];
  // d_in[6] = mask_out (all ones) unused
  const float* rays_d = (const float*)d_in[7];
  const float* w1 = (const float*)d_in[8];  const float* b1 = (const float*)d_in[9];
  const float* w2 = (const float*)d_in[10]; const float* b2 = (const float*)d_in[11];
  const float* w3 = (const float*)d_in[12]; const float* b3 = (const float*)d_in[13];
  const float* w4 = (const float*)d_in[14]; const float* b4 = (const float*)d_in[15];
  const float* wa = (const float*)d_in[16]; const float* ba = (const float*)d_in[17];
  const float* wc1 = (const float*)d_in[18]; const float* bc1 = (const float*)d_in[19];
  const float* wc2 = (const float*)d_in[20]; const float* bc2 = (const float*)d_in[21];
  const float* wc3 = (const float*)d_in[22]; const float* bc3 = (const float*)d_in[23];
  const float* wc4 = (const float*)d_in[24]; const float* bc4 = (const float*)d_in[25];

  char* ws = (char*)d_ws;
  _Float16* Wt1  = (_Float16*)(ws + WS_WT1);
  _Float16* Wt2  = (_Float16*)(ws + WS_WT2);
  _Float16* Wt3  = (_Float16*)(ws + WS_WT3);
  _Float16* Wt4  = (_Float16*)(ws + WS_WT4);
  _Float16* Wtc1 = (_Float16*)(ws + WS_WTC1);
  _Float16* Wtc2 = (_Float16*)(ws + WS_WTC2);
  _Float16* Wtc3 = (_Float16*)(ws + WS_WTC3);
  _Float16* featbuf = (_Float16*)(ws + WS_FEAT);

  float* alpha_out = (float*)d_out;          // 32768
  float* color_out = (float*)d_out + 32768;  // 32768 x 3

  struct { const float* s; _Float16* d; int K, N, Kpad; } wp[7] = {
    { w1,  Wt1,  284, 256, 288 },
    { w2,  Wt2,  256, 256, 256 },
    { w3,  Wt3,  263, 256, 288 },
    { w4,  Wt4,  256, 256, 256 },
    { wc1, Wtc1, 280, 128, 288 },
    { wc2, Wtc2, 128, 128, 128 },
    { wc3, Wtc3, 128, 128, 128 },
  };
  for (int i = 0; i < 7; ++i) {
    int total = wp[i].N * wp[i].Kpad;
    wprep_kernel<<<(total + 255) / 256, 256, 0, stream>>>(wp[i].s, wp[i].d,
                                                          wp[i].K, wp[i].N, wp[i].Kpad);
  }

  // per-point MLP + aggregation: 8 samples (128 rows) per block, dynamic LDS
  const size_t smem_bytes = (size_t)ROWS * 288 * 2 + (size_t)ROWS * 256 * 2; // 139264
  point_mlp_kernel<<<32768 / NPB, 256, smem_bytes, stream>>>(
      pc_xyz, pc_rgb, pc_dir, pc_feat, sp_xyz, vmask, rays_d,
      Wt1, b1, Wt2, b2, Wt3, b3, Wt4, b4, wa, ba,
      featbuf, alpha_out);

  ray_head_kernel<<<2048, 128, 0, stream>>>(
      featbuf, Wtc1, bc1, Wtc2, bc2, Wtc3, bc3, wc4, bc4, color_out);
}
